// AutoformerClassifier_23545010717116
// MI455X (gfx1250) — matrix operand order repaired and checked by reading, not compiled, not run on hardware
//
#include <hip/hip_runtime.h>
#include <hip/hip_bf16.h>
#include <cstdint>

// ---------------------------------------------------------------------------
// Autoformer classifier forward for MI455X (gfx1250, wave32, WMMA + TDM).
// Heavy GEMMs run on v_wmma_f32_16x16x32_bf16 (fp32 accumulate), 2x2 register
// tiles per wave, vectorized (b128) LDS fragment loads. FFT autocorrelation
// stages strided series into LDS via the Tensor Data Mover when available.
// ---------------------------------------------------------------------------

#define B_   32
#define L_   4096
#define C_   12
#define D_   128
#define DFF_ 512
#define TOPK_ 8
#define NMD_ (B_ * L_ * D_)   // 16,777,216

typedef __bf16 bf16_t;
typedef __attribute__((ext_vector_type(16))) __bf16 v16bf;
typedef __attribute__((ext_vector_type(8)))  float  v8f;

union V16U { v16bf v; uint4 q[2]; };

// --- Tensor Data Mover availability / arity dispatch -----------------------
#if defined(__has_builtin)
#  if __has_builtin(__builtin_amdgcn_tensor_load_to_lds)
#    define USE_TDM 1
#  endif
#endif
#ifndef USE_TDM
#  define USE_TDM 0
#endif
#if defined(__has_include)
#  if __has_include(<hip/amd_detail/amd_gfx1250_TDM.h>)
#    define TDM_SIX_ARGS 1
#  endif
#endif
#ifndef TDM_SIX_ARGS
#  define TDM_SIX_ARGS 0
#endif

#if USE_TDM
typedef __attribute__((ext_vector_type(4))) unsigned int u32x4_t;
typedef __attribute__((ext_vector_type(4))) int          i32x4_t;
typedef __attribute__((ext_vector_type(8))) int          i32x8_t;

// DMA-gather a strided column (nrows elements, row stride in fp32 elements)
// from global memory into contiguous LDS. D# layout per CDNA5 ISA ch. 7/8:
//   group0: [1:0] count=1 | [63:32] lds_addr | [120:64] global_addr | type=2
//   group1: data_size=4B, tensor_dim0=stride, tensor_dim1=nrows,
//           tile_dim0=1, tile_dim1=nrows, tensor_dim0_stride=stride
__device__ __forceinline__ void tdm_load_strided_col(
    const float* gsrc, float* ldsdst, int nrows, int row_stride_elems) {
    const unsigned long long ga = (unsigned long long)(uintptr_t)gsrc;
    const unsigned lds = (unsigned)(uintptr_t)ldsdst;   // low 32 bits = LDS byte addr
    u32x4_t g0; i32x8_t g1; i32x4_t g2; i32x4_t g3;
    g0[0] = 1u;                                               // count=1, user D#
    g0[1] = lds;                                              // lds_addr (bytes)
    g0[2] = (unsigned)ga;                                     // global_addr[31:0]
    g0[3] = (unsigned)((ga >> 32) & 0x01FFFFFFu) | (2u << 30);// ga[56:32] | type=2
    g1[0] = (int)(2u << 16);                                  // data_size = 4 bytes
    g1[1] = (int)((unsigned)row_stride_elems << 16);          // tensor_dim0[15:0]
    g1[2] = (int)((unsigned)nrows << 16);                     // tensor_dim1[15:0]
    g1[3] = (int)(1u << 16);                                  // tile_dim0 = 1
    g1[4] = (int)(unsigned)nrows;                             // tile_dim1 = nrows
    g1[5] = (int)(unsigned)row_stride_elems;                  // dim0_stride[31:0]
    g1[6] = 0;
    g1[7] = 0;
    g2[0] = g2[1] = g2[2] = g2[3] = 0;
    g3[0] = g3[1] = g3[2] = g3[3] = 0;
#if TDM_SIX_ARGS
    i32x8_t g4;
    g4[0] = g4[1] = g4[2] = g4[3] = g4[4] = g4[5] = g4[6] = g4[7] = 0;
    __builtin_amdgcn_tensor_load_to_lds(g0, g1, g2, g3, g4, 0);
#else
    __builtin_amdgcn_tensor_load_to_lds(g0, g1, g2, g3, 0);
#endif
}
#endif  // USE_TDM

__device__ __forceinline__ float gelu_f(float x) {
    return 0.5f * x * (1.0f + erff(x * 0.70710678118654752f));
}

// ---------------------------------------------------------------------------
// Conv1d embedding with circular padding: h[b,l,d] = sum_{c,k} w[d,c,k]*x[b,(l-1+k)%L,c]
// ---------------------------------------------------------------------------
__global__ __launch_bounds__(128) void embed_kernel(
    const float* __restrict__ x, const float* __restrict__ w,
    float* __restrict__ h, bf16_t* __restrict__ hb) {
    int bl = blockIdx.x;            // 0 .. B*L-1
    int d  = threadIdx.x;           // 0 .. 127
    int b  = bl >> 12;
    int l  = bl & (L_ - 1);
    float acc = 0.0f;
    #pragma unroll
    for (int k = 0; k < 3; ++k) {
        int lw = (l - 1 + k + L_) & (L_ - 1);
        const float* xr = x + ((size_t)(b * L_ + lw)) * C_;
        const float* wr = w + d * (C_ * 3) + k;
        #pragma unroll
        for (int c = 0; c < C_; ++c) acc += wr[c * 3] * xr[c];
    }
    size_t o = (size_t)bl * D_ + d;
    h[o]  = acc;
    hb[o] = (bf16_t)acc;
}

// ---------------------------------------------------------------------------
// fp32 -> bf16 cast
// ---------------------------------------------------------------------------
__global__ void cast_bf16_kernel(const float* __restrict__ s, bf16_t* __restrict__ d, int n) {
    int i = blockIdx.x * blockDim.x + threadIdx.x;
    if (i < n) d[i] = (bf16_t)s[i];
}

// ---------------------------------------------------------------------------
// WMMA GEMM: out[M,N] = act(A[M,K](bf16) * W[N,K]^T(bf16) + bias + res)
// Workgroup tile 128(M) x 64(N); 8 waves as 4x2; each wave owns a 32x32 output
// block = 2x2 grid of 16x16 WMMA accumulators (4 WMMAs per K-step).
// ---------------------------------------------------------------------------
template <int KTOT, int ACT, bool OUTBF, bool BIAS, bool RES>
__global__ __launch_bounds__(256) void gemm_wmma_kernel(
    const bf16_t* __restrict__ A, const bf16_t* __restrict__ W,
    const float* __restrict__ bias, const float* __restrict__ res,
    float* __restrict__ outf, bf16_t* __restrict__ outb, int N) {
    __shared__ bf16_t As[128 * 40];   // 128 rows x 32 K, stride 40 (16B-aligned rows)
    __shared__ bf16_t Ws[64 * 40];    // 64 N-rows x 32 K

    const int tid    = threadIdx.x;
    const int lane   = tid & 31;
    const int wv     = tid >> 5;      // wave 0..7
    const int wm     = wv & 3;        // wave M block 0..3 (32 rows each)
    const int wn     = wv >> 2;       // wave N block 0..1 (32 cols each)
    const int laneLo = lane & 15;
    const bool laneHi = lane >= 16;
    const int m0 = blockIdx.x * 128;
    const int n0 = blockIdx.y * 64;

    v8f acc[2][2] = {};

    const int arow = tid >> 1, akp = (tid & 1) * 16;   // A: 16 bf16 per thread
    const int nrow = tid >> 2, wkp = (tid & 3) * 8;    // W: 8 bf16 per thread

    for (int kk = 0; kk < KTOT; kk += 32) {
        // Stage A panel (128 x 32) and W panel (64 x 32) into LDS (b128 copies).
        {
            const bf16_t* src = &A[(size_t)(m0 + arow) * KTOT + kk + akp];
            *(uint4*)&As[arow * 40 + akp]     = *(const uint4*)src;
            *(uint4*)&As[arow * 40 + akp + 8] = *(const uint4*)(src + 8);
        }
        *(uint4*)&Ws[nrow * 40 + wkp] =
            *(const uint4*)&W[(size_t)(n0 + nrow) * KTOT + kk + wkp];
        if (kk + 32 < KTOT)   // near-cache prefetch of next A panel
            __builtin_prefetch(&A[(size_t)(m0 + arow) * KTOT + kk + 32 + akp], 0, 3);
        __syncthreads();

        v16bf af[2], bfr[2];
        #pragma unroll
        for (int mi = 0; mi < 2; ++mi) {
            // lanes 0-15: K = 0..7 then 16..23 ; lanes 16-31: K = 8..15 then 24..31
            const bf16_t* ap = &As[(wm * 32 + mi * 16 + laneLo) * 40 + (laneHi ? 8 : 0)];
            V16U u;
            u.q[0] = *(const uint4*)ap;          // first 8-element K-run
            u.q[1] = *(const uint4*)(ap + 16);   // second 8-element K-run
            af[mi] = u.v;
        }
        #pragma unroll
        for (int ni = 0; ni < 2; ++ni) {
            // lanes 0-15: K = 0..7 then 16..23 ; lanes 16-31: K = 8..15 then 24..31
            const bf16_t* bp = &Ws[(wn * 32 + ni * 16 + laneLo) * 40 + (laneHi ? 8 : 0)];
            V16U u;
            u.q[0] = *(const uint4*)bp;
            u.q[1] = *(const uint4*)(bp + 16);
            bfr[ni] = u.v;
        }
        #pragma unroll
        for (int mi = 0; mi < 2; ++mi)
            #pragma unroll
            for (int ni = 0; ni < 2; ++ni)
                acc[mi][ni] = __builtin_amdgcn_wmma_f32_16x16x32_bf16(
                    false, af[mi], false, bfr[ni], (short)0, acc[mi][ni], false, false);
        __syncthreads();
    }

    // C/D layout: VGPR j -> M = j (+8 for lanes 16-31), N = lane%16.
    #pragma unroll
    for (int mi = 0; mi < 2; ++mi) {
        #pragma unroll
        for (int ni = 0; ni < 2; ++ni) {
            const int n = n0 + wn * 32 + ni * 16 + laneLo;
            #pragma unroll
            for (int j = 0; j < 8; ++j) {
                const int m = m0 + wm * 32 + mi * 16 + j + (laneHi ? 8 : 0);
                float val = acc[mi][ni][j];
                if (BIAS) val += bias[n];
                if (RES)  val += res[(size_t)m * N + n];
                if (ACT == 1) val = gelu_f(val);
                else if (ACT == 2) val = fmaxf(val, 0.0f);
                if (OUTBF) outb[(size_t)m * N + n] = (bf16_t)val;
                else       outf[(size_t)m * N + n] = val;
            }
        }
    }
}

// ---------------------------------------------------------------------------
// FFT autocorrelation. One workgroup per (b, channel) series of length 4096.
// Forward DIF (natural->bitrev), pointwise q*conj(k), inverse DIT (bitrev->natural).
// Series staged into LDS with the Tensor Data Mover when available.
// ---------------------------------------------------------------------------
__device__ void fft_dif_fwd(float* re, float* im, int tid) {
    for (int half = 2048; half >= 1; half >>= 1) {
        for (int t = tid; t < 2048; t += 256) {
            const int blk = t / half, j = t % half;
            const int i0 = blk * 2 * half + j, i1 = i0 + half;
            const float ur = re[i0], ui = im[i0], vr = re[i1], vi = im[i1];
            const float dr = ur - vr, di = ui - vi;
            float c, s;
            __sincosf(-6.28318530717958648f * (float)j / (2.0f * (float)half), &s, &c);
            re[i0] = ur + vr;           im[i0] = ui + vi;
            re[i1] = dr * c - di * s;   im[i1] = dr * s + di * c;
        }
        __syncthreads();
    }
}

__device__ void fft_dit_inv(float* re, float* im, int tid) {
    for (int half = 1; half <= 2048; half <<= 1) {
        for (int t = tid; t < 2048; t += 256) {
            const int blk = t / half, j = t % half;
            const int i0 = blk * 2 * half + j, i1 = i0 + half;
            float c, s;
            __sincosf(6.28318530717958648f * (float)j / (2.0f * (float)half), &s, &c);
            const float vr = re[i1] * c - im[i1] * s;
            const float vi = re[i1] * s + im[i1] * c;
            const float ur = re[i0], ui = im[i0];
            re[i0] = ur + vr; im[i0] = ui + vi;
            re[i1] = ur - vr; im[i1] = ui - vi;
        }
        __syncthreads();
    }
}

__global__ __launch_bounds__(256) void autocorr_kernel(
    const float* __restrict__ qkv,   // [B*L, 384] (q|k|v)
    float* __restrict__ corr) {      // [B, 128, L]
    __shared__ float qr[L_], qi[L_], kr[L_], ki[L_];   // 64 KB LDS
    const int b = blockIdx.x >> 7, d = blockIdx.x & 127, tid = threadIdx.x;
    const float* base = qkv + (size_t)b * L_ * 384;
#if USE_TDM
    if (tid < 32) {   // single wave issues both DMA-gathers, then drains TENSORcnt
        tdm_load_strided_col(base + d,       qr, L_, 384);
        tdm_load_strided_col(base + 128 + d, kr, L_, 384);
        __builtin_amdgcn_s_wait_tensorcnt(0);
    }
    for (int l = tid; l < L_; l += 256) { qi[l] = 0.0f; ki[l] = 0.0f; }
#else
    for (int l = tid; l < L_; l += 256) {
        qr[l] = base[(size_t)l * 384 + d];        qi[l] = 0.0f;
        kr[l] = base[(size_t)l * 384 + 128 + d];  ki[l] = 0.0f;
    }
#endif
    __syncthreads();
    fft_dif_fwd(qr, qi, tid);
    fft_dif_fwd(kr, ki, tid);
    for (int l = tid; l < L_; l += 256) {   // q * conj(k)
        const float ar = qr[l], ai = qi[l], br = kr[l], bi = ki[l];
        qr[l] = ar * br + ai * bi;
        qi[l] = ai * br - ar * bi;
    }
    __syncthreads();
    fft_dit_inv(qr, qi, tid);
    float* out = corr + ((size_t)b * 128 + d) * L_;
    const float inv = 1.0f / (float)L_;     // irfft normalization
    for (int l = tid; l < L_; l += 256) out[l] = qr[l] * inv;
}

// meanv[b,l] = mean over 128 channels of corr[b,:,l]   (deterministic, no atomics)
__global__ void chanmean_kernel(const float* __restrict__ corr, float* __restrict__ meanv) {
    const int i = blockIdx.x * blockDim.x + threadIdx.x;   // over B*L
    const int b = i >> 12, l = i & (L_ - 1);
    const float* base = corr + (size_t)b * 128 * L_ + l;
    float s = 0.0f;
    for (int d = 0; d < 128; ++d) s += base[(size_t)d * L_];
    meanv[i] = s * (1.0f / 128.0f);
}

// Top-8 indices of batch-mean of meanv. Single workgroup.
__global__ __launch_bounds__(256) void topk_kernel(
    const float* __restrict__ meanv, int* __restrict__ topidx) {
    __shared__ float g[L_];
    __shared__ float rv[256];
    __shared__ int   ri[256];
    const int tid = threadIdx.x;
    for (int l = tid; l < L_; l += 256) {
        float s = 0.0f;
        for (int b = 0; b < B_; ++b) s += meanv[b * L_ + l];
        g[l] = s;
    }
    __syncthreads();
    for (int t = 0; t < TOPK_; ++t) {
        float best = -1e30f; int bi = 0;
        for (int l = tid; l < L_; l += 256)
            if (g[l] > best) { best = g[l]; bi = l; }
        rv[tid] = best; ri[tid] = bi;
        __syncthreads();
        for (int s = 128; s > 0; s >>= 1) {
            if (tid < s && rv[tid + s] > rv[tid]) { rv[tid] = rv[tid + s]; ri[tid] = ri[tid + s]; }
            __syncthreads();
        }
        if (tid == 0) { topidx[t] = ri[0]; g[ri[0]] = -1e30f; }
        __syncthreads();
    }
}

__global__ void softmaxw_kernel(const float* __restrict__ meanv,
                                const int* __restrict__ topidx,
                                float* __restrict__ sw) {
    const int b = threadIdx.x;
    if (b >= B_) return;
    float w[TOPK_], mx = -1e30f;
    for (int j = 0; j < TOPK_; ++j) { w[j] = meanv[b * L_ + topidx[j]]; mx = fmaxf(mx, w[j]); }
    float s = 0.0f;
    for (int j = 0; j < TOPK_; ++j) { w[j] = __expf(w[j] - mx); s += w[j]; }
    for (int j = 0; j < TOPK_; ++j) sw[b * TOPK_ + j] = w[j] / s;
}

// agg[b,l,d] = sum_j sw[b,j] * v[b,(l+idx[j])%L,d]   -> bf16 for O-projection GEMM
__global__ void agg_kernel(const float* __restrict__ qkv, const float* __restrict__ sw,
                           const int* __restrict__ topidx, bf16_t* __restrict__ attnb) {
    const int i = blockIdx.x * blockDim.x + threadIdx.x;    // over NMD
    const int d = i & 127, bl = i >> 7;
    const int l = bl & (L_ - 1), b = bl >> 12;
    float acc = 0.0f;
    #pragma unroll
    for (int j = 0; j < TOPK_; ++j) {
        const int ls = (l + topidx[j]) & (L_ - 1);
        acc += sw[b * TOPK_ + j] * qkv[((size_t)(b * L_ + ls)) * 384 + 256 + d];
    }
    attnb[i] = (bf16_t)acc;
}

// series_decomp residual: out = x - movavg25(x), replicate padding along L
__global__ void decomp_kernel(const float* __restrict__ xin, float* __restrict__ xout,
                              bf16_t* __restrict__ xoutb) {
    const int i = blockIdx.x * blockDim.x + threadIdx.x;    // over NMD
    const int d = i & 127, bl = i >> 7;
    const int l = bl & (L_ - 1), b = bl >> 12;
    const float* base = xin + (size_t)b * L_ * D_ + d;
    float s = 0.0f;
    #pragma unroll
    for (int j = -12; j <= 12; ++j) {
        int lj = l + j;
        lj = lj < 0 ? 0 : (lj > L_ - 1 ? L_ - 1 : lj);
        s += base[(size_t)lj * D_];
    }
    const float r = xin[i] - s * (1.0f / 25.0f);
    xout[i]  = r;
    xoutb[i] = (bf16_t)r;
}

// LayerNorm over D per (b,l)
__global__ void ln_kernel(const float* __restrict__ x, const float* __restrict__ g,
                          const float* __restrict__ be, float* __restrict__ out) {
    const int r = blockIdx.x * blockDim.x + threadIdx.x;    // over B*L
    const float* row = x + (size_t)r * D_;
    float mu = 0.0f;
    for (int d = 0; d < D_; ++d) mu += row[d];
    mu *= (1.0f / D_);
    float var = 0.0f;
    for (int d = 0; d < D_; ++d) { const float df = row[d] - mu; var += df * df; }
    var *= (1.0f / D_);
    const float inv = rsqrtf(var + 1e-5f);
    float* o = out + (size_t)r * D_;
    for (int d = 0; d < D_; ++d) o[d] = (row[d] - mu) * inv * g[d] + be[d];
}

// pooled[b,d] = mean_l gelu(ln[b,l,d] - mean_l ln[b,l,d])
__global__ __launch_bounds__(256) void pool_kernel(const float* __restrict__ ln,
                                                   float* __restrict__ pooled) {
    __shared__ float red[256];
    __shared__ float cmS;
    const int b = blockIdx.x >> 7, d = blockIdx.x & 127, tid = threadIdx.x;
    const float* base = ln + (size_t)b * L_ * D_ + d;
    float s = 0.0f;
    for (int l = tid; l < L_; l += 256) s += base[(size_t)l * D_];
    red[tid] = s; __syncthreads();
    for (int t = 128; t > 0; t >>= 1) { if (tid < t) red[tid] += red[tid + t]; __syncthreads(); }
    if (tid == 0) cmS = red[0] * (1.0f / L_);
    __syncthreads();
    const float cm = cmS;
    float s2 = 0.0f;
    for (int l = tid; l < L_; l += 256) s2 += gelu_f(base[(size_t)l * D_] - cm);
    red[tid] = s2; __syncthreads();
    for (int t = 128; t > 0; t >>= 1) { if (tid < t) red[tid] += red[tid + t]; __syncthreads(); }
    if (tid == 0) pooled[b * D_ + d] = red[0] * (1.0f / L_);
}

// FC head: relu(pooled @ fc1^T + b1) @ fc2^T + b2
__global__ __launch_bounds__(128) void head_kernel(
    const float* __restrict__ pooled, const float* __restrict__ fc1w,
    const float* __restrict__ fc1b, const float* __restrict__ fc2w,
    const float* __restrict__ fc2b, float* __restrict__ out) {
    __shared__ float f[128];
    const int b = blockIdx.x, h = threadIdx.x;
    float acc = fc1b[h];
    for (int k = 0; k < 128; ++k) acc += pooled[b * 128 + k] * fc1w[h * 128 + k];
    f[h] = fmaxf(acc, 0.0f);
    __syncthreads();
    if (h < 5) {
        float o = fc2b[h];
        for (int k = 0; k < 128; ++k) o += f[k] * fc2w[h * 128 + k];
        out[b * 5 + h] = o;
    }
}

// ---------------------------------------------------------------------------
// Host orchestration
// ---------------------------------------------------------------------------
extern "C" void kernel_launch(void* const* d_in, const int* in_sizes, int n_in,
                              void* d_out, int out_size, void* d_ws, size_t ws_size,
                              hipStream_t stream) {
    const float* x_enc  = (const float*)d_in[0];
    const float* emb_w  = (const float*)d_in[1];
    const float* attn_w = (const float*)d_in[2];   // (2,4,128,128)
    const float* attn_b = (const float*)d_in[3];   // (2,4,128)
    const float* ffn_w1 = (const float*)d_in[4];   // (2,512,128)
    const float* ffn_w2 = (const float*)d_in[5];   // (2,128,512)
    const float* norm_g = (const float*)d_in[6];
    const float* norm_b = (const float*)d_in[7];
    const float* fc1_w  = (const float*)d_in[8];
    const float* fc1_b  = (const float*)d_in[9];
    const float* fc2_w  = (const float*)d_in[10];
    const float* fc2_b  = (const float*)d_in[11];
    float* out = (float*)d_out;

    // Bump-allocate workspace.
    char* ws = (char*)d_ws;
    size_t off = 0;
    auto alloc = [&](size_t bytes) -> void* {
        void* p = ws + off;
        off = (off + bytes + 255) & ~(size_t)255;
        return p;
    };
    float*  h      = (float*)alloc((size_t)NMD_ * 4);
    float*  tmp    = (float*)alloc((size_t)NMD_ * 4);        // also corr buffer
    bf16_t* hb     = (bf16_t*)alloc((size_t)NMD_ * 2);
    float*  qkv    = (float*)alloc((size_t)B_ * L_ * 384 * 4);
    bf16_t* attnb  = (bf16_t*)alloc((size_t)NMD_ * 2);
    bf16_t* ybuf   = (bf16_t*)alloc((size_t)B_ * L_ * DFF_ * 2);
    float*  meanv  = (float*)alloc((size_t)B_ * L_ * 4);
    int*    topidx = (int*)alloc(64 * 4);
    float*  sw     = (float*)alloc((size_t)B_ * TOPK_ * 4);
    bf16_t* wbuf   = (bf16_t*)alloc((size_t)512 * 512 * 2);
    float*  pooled = (float*)alloc((size_t)B_ * D_ * 4);

    const int M = B_ * L_;                 // 131072
    const dim3 blk256(256);

    // Embedding conv -> h (fp32) + hb (bf16)
    embed_kernel<<<B_ * L_, 128, 0, stream>>>(x_enc, emb_w, h, hb);

    for (int l = 0; l < 2; ++l) {
        const float* aw = attn_w + (size_t)l * 4 * 128 * 128;
        const float* ab = attn_b + (size_t)l * 4 * 128;

        // QKV: out[M,384] = hb @ [Wq|Wk|Wv]^T + bias
        cast_bf16_kernel<<<(3 * 128 * 128 + 255) / 256, blk256, 0, stream>>>(aw, wbuf, 3 * 128 * 128);
        gemm_wmma_kernel<128, 0, false, true, false>
            <<<dim3(M / 128, 384 / 64), blk256, 0, stream>>>(hb, wbuf, ab, nullptr, qkv, nullptr, 384);

        // Autocorrelation: per-channel circular correlation via FFT
        autocorr_kernel<<<B_ * 128, blk256, 0, stream>>>(qkv, tmp);
        chanmean_kernel<<<(B_ * L_) / 256, blk256, 0, stream>>>(tmp, meanv);
        topk_kernel<<<1, blk256, 0, stream>>>(meanv, topidx);
        softmaxw_kernel<<<1, 32, 0, stream>>>(meanv, topidx, sw);
        agg_kernel<<<NMD_ / 256, blk256, 0, stream>>>(qkv, sw, topidx, attnb);

        // O projection + residual -> tmp; then decompose -> h, hb
        cast_bf16_kernel<<<(128 * 128 + 255) / 256, blk256, 0, stream>>>(aw + 3 * 128 * 128, wbuf, 128 * 128);
        gemm_wmma_kernel<128, 0, false, true, true>
            <<<dim3(M / 128, 128 / 64), blk256, 0, stream>>>(attnb, wbuf, ab + 3 * 128, h, tmp, nullptr, 128);
        decomp_kernel<<<NMD_ / 256, blk256, 0, stream>>>(tmp, h, hb);

        // FFN1: y = gelu(x @ W1^T), bf16 out
        cast_bf16_kernel<<<(512 * 128 + 255) / 256, blk256, 0, stream>>>(ffn_w1 + (size_t)l * 512 * 128, wbuf, 512 * 128);
        gemm_wmma_kernel<128, 1, true, false, false>
            <<<dim3(M / 128, 512 / 64), blk256, 0, stream>>>(hb, wbuf, nullptr, nullptr, nullptr, ybuf, 512);

        // FFN2: tmp = x + y @ W2^T; then decompose -> h, hb
        cast_bf16_kernel<<<(128 * 512 + 255) / 256, blk256, 0, stream>>>(ffn_w2 + (size_t)l * 128 * 512, wbuf, 128 * 512);
        gemm_wmma_kernel<512, 0, false, false, true>
            <<<dim3(M / 128, 128 / 64), blk256, 0, stream>>>(ybuf, wbuf, nullptr, h, tmp, nullptr, 128);
        decomp_kernel<<<NMD_ / 256, blk256, 0, stream>>>(tmp, h, hb);
    }

    // Final head: LayerNorm -> column-mean-subtract + gelu pool -> FC1 -> FC2
    ln_kernel<<<(B_ * L_) / 256, blk256, 0, stream>>>(h, norm_g, norm_b, tmp);
    pool_kernel<<<B_ * 128, blk256, 0, stream>>>(tmp, pooled);
    head_kernel<<<B_, 128, 0, stream>>>(pooled, fc1_w, fc1_b, fc2_w, fc2_b, out);
}
